// TransMLAAbsorber_47304769798936
// MI455X (gfx1250) — matrix operand order repaired and checked by reading, not compiled, not run on hardware
//
#include <hip/hip_runtime.h>

// ---------------- problem constants ----------------
#define B_    2
#define HQ_   32
#define HKV_  8
#define QL_   1024
#define SL_   4096
#define RANK_ 512
#define ROPE_ 64
#define HALF_ 64
#define DH_   128

// ---------------- workspace layout (bytes) ----------------
#define OFF_MABS   0u                                   // 64*512 f32      = 131072
#define OFF_WUVT   131072u                              // 128*512 bf16    = 131072
#define OFF_QABS   262144u                              // 2*32*1024*512 bf16 = 67108864
#define OFF_QROPE  67371008u                            // 2*32*1024*64 bf16  = 8388608
#define OFF_CKV    75759616u                            // 2*4096*512 bf16    = 8388608
#define OFF_CKVT   84148224u                            // 2*512*4096 bf16    = 8388608
#define OFF_KROPE  92536832u                            // 2*8*4096*64 bf16   = 8388608

typedef __attribute__((ext_vector_type(16))) __bf16 v16bf;
typedef __attribute__((ext_vector_type(8)))  __bf16 v8bf;
typedef __attribute__((ext_vector_type(8)))  float  v8f;

union V16U { v16bf v; v8bf h[2]; };

// sched_group_barrier masks (LLVM AMDGPU IGroupLP)
#define SG_MFMA    0x008   // MFMA / WMMA
#define SG_VMEM_RD 0x020
#define SG_DS_RD   0x100

__device__ __forceinline__ unsigned short f2bf(float f) {
    unsigned int u = __float_as_uint(f);
    u += 0x7FFFu + ((u >> 16) & 1u);          // round to nearest even
    return (unsigned short)(u >> 16);
}

__device__ __forceinline__ v8bf ld8(const unsigned short* p) {
    return *reinterpret_cast<const v8bf*>(p);
}

// A fragment (16xK tile, 16-bit): row-major source, row m=lane&15,
// K-runs at kbase+(lane>>4)*8 and +16 (ISA 7.12.2, 16-bit A 16x32).
__device__ __forceinline__ v16bf load_a(const unsigned short* base, int ld, int kbase, int lane) {
    const int m  = lane & 15;
    const int kh = (lane >> 4) * 8;
    const unsigned short* p = base + m * ld + kbase + kh;
    V16U u; u.h[0] = ld8(p); u.h[1] = ld8(p + 16);
    return u.v;
}

// B fragment (Kx16 tile, 16-bit) from B^T stored row-major (n-major):
// lane holds column n=nbase+(lane&15), K-runs of 8 at kbase+(lane>>4)*8 and +16.
__device__ __forceinline__ v16bf load_bT(const unsigned short* baseT, int ld, int nbase, int kbase, int lane) {
    const int n = nbase + (lane & 15);
    const int k = kbase + ((lane >> 4) << 3);
    const unsigned short* p = baseT + n * ld + k;
    V16U u; u.h[0] = ld8(p); u.h[1] = ld8(p + 16);
    return u.v;
}

__device__ __forceinline__ v8f wmma_bf16(v16bf a, v16bf b, v8f c) {
    return __builtin_amdgcn_wmma_f32_16x16x32_bf16(false, a, false, b, (short)0, c, false, false);
}

// pre-swizzled A-fragment cache in LDS: frag f, half h, lane -> 16B
__device__ __forceinline__ v16bf lda_lds(const char* sA, int f, int lane) {
    V16U u;
    u.h[0] = *reinterpret_cast<const v8bf*>(sA + ((f * 2 + 0) * 32 + lane) * 16);
    u.h[1] = *reinterpret_cast<const v8bf*>(sA + ((f * 2 + 1) * 32 + lane) * 16);
    return u.v;
}

__device__ __forceinline__ void sta_lds(char* sA, int f, int lane, v16bf a) {
    V16U u; u.v = a;
    *reinterpret_cast<v8bf*>(sA + ((f * 2 + 0) * 32 + lane) * 16) = u.h[0];
    *reinterpret_cast<v8bf*>(sA + ((f * 2 + 1) * 32 + lane) * 16) = u.h[1];
}

// ---------------- prep kernels ----------------

// Mabs[i][j] = sum_k U_l[k][i] * W_UK[k][j]   (U_l^T @ W_UK, 64x512, f32)
__global__ __launch_bounds__(256) void k_mabs(const float* __restrict__ U_l,
                                              const float* __restrict__ W_UK,
                                              float* __restrict__ Mabs) {
    int idx = blockIdx.x * 256 + threadIdx.x;       // 0..32767
    int i = idx >> 9, j = idx & 511;
    float a = 0.f;
    #pragma unroll 8
    for (int k = 0; k < 64; ++k) a += U_l[k * 64 + i] * W_UK[k * 512 + j];
    Mabs[idx] = a;
}

// WUVT[d][r] = bf16(W_UV[r][d])   (128x512)
__global__ __launch_bounds__(256) void k_wuvt(const float* __restrict__ W_UV,
                                              unsigned short* __restrict__ WUVT) {
    int idx = blockIdx.x * 256 + threadIdx.x;       // 0..65535
    int d = idx >> 9, r = idx & 511;
    WUVT[idx] = f2bf(W_UV[r * DH_ + d]);
}

// Ckv -> bf16 copy + transposed bf16 copy CkvT[b][r][s]
__global__ __launch_bounds__(256) void k_ckv(const float* __restrict__ C,
                                             unsigned short* __restrict__ Cb,
                                             unsigned short* __restrict__ CTb) {
    size_t idx = (size_t)blockIdx.x * 256 + threadIdx.x;   // N = 2*4096*512
    int b   = (int)(idx / ((size_t)SL_ * RANK_));
    int rem = (int)(idx % ((size_t)SL_ * RANK_));
    int s = rem >> 9, r = rem & 511;
    unsigned short h = f2bf(C[idx]);
    Cb[idx] = h;
    CTb[(size_t)b * RANK_ * SL_ + (size_t)r * SL_ + s] = h;
}

// Krope -> bf16
__global__ __launch_bounds__(256) void k_krope(const float* __restrict__ K,
                                               unsigned short* __restrict__ Kb) {
    size_t idx = (size_t)blockIdx.x * 256 + threadIdx.x;   // N = 2*8*4096*64
    Kb[idx] = f2bf(K[idx]);
}

// Per q-row: Q_rope = Qx @ U_l^T ; Q_abs = Qy @ Mabs  (both -> bf16)
__global__ __launch_bounds__(128) void k_prepq(const float* __restrict__ Q,
                                               const float* __restrict__ U_l,
                                               const float* __restrict__ Mabs,
                                               unsigned short* __restrict__ Qabs,
                                               unsigned short* __restrict__ Qrope) {
    const int row = blockIdx.x;                 // 0 .. B*HQ*QL-1
    const int t = threadIdx.x;
    __shared__ float qs[DH_];
    qs[t] = Q[(size_t)row * DH_ + t];
    __syncthreads();
    if (t < 64) {
        float a = 0.f;
        #pragma unroll 8
        for (int k = 0; k < 64; ++k) a += qs[k] * U_l[t * 64 + k];
        Qrope[(size_t)row * ROPE_ + t] = f2bf(a);
    }
    #pragma unroll
    for (int j = t; j < RANK_; j += 128) {
        float a = 0.f;
        #pragma unroll 8
        for (int i = 0; i < 64; ++i) a += qs[64 + i] * Mabs[i * RANK_ + j];
        Qabs[(size_t)row * RANK_ + j] = f2bf(a);
    }
}

// ---------------- main flash-attention kernel ----------------
// grid: B*HQ*(QL/16) = 4096 blocks, 256 threads (8 wave32).
// Q A-fragments live in LDS (identical across waves); QK and PV use depth-4
// rotating B buffers + depth-2 rotating A buffers with sched_group_barrier
// pinning so each WMMA has >=3 iterations of load latency cover.
__global__ __launch_bounds__(256) void mla_attn(const unsigned short* __restrict__ Qabs,
                                                const unsigned short* __restrict__ Qrope,
                                                const unsigned short* __restrict__ Ckv,
                                                const unsigned short* __restrict__ CkvT,
                                                const unsigned short* __restrict__ Krope,
                                                const unsigned short* __restrict__ WUVT,
                                                float* __restrict__ Out) {
    __shared__ __align__(16) char sA[18 * 2 * 32 * 16];   // 18 pre-swizzled A frags (18KB)
    __shared__ float          sS[16 * 132];               // fp32 scores 16x128 (stride 132)
    __shared__ unsigned short sP[16 * 136];               // bf16 P 16x128 (stride 136)
    __shared__ unsigned short sO[16 * 520];               // bf16 out_latent 16x512 (stride 520)
    __shared__ float sM[16], sL[16], sScale[16];

    const int t    = threadIdx.x;
    const int lane = t & 31;
    const int w    = t >> 5;                          // wave id 0..7
    const int bid  = blockIdx.x;
    const int qt = bid & 63, h = (bid >> 6) & 31, b = bid >> 11;
    const int q0 = qt << 4;
    const int hk = h >> 2;                            // HQ/HKV = 4

    const size_t qoff = (size_t)(b * HQ_ + h) * QL_ + q0;
    const unsigned short* Qa  = Qabs  + qoff * RANK_;
    const unsigned short* Qr  = Qrope + qoff * ROPE_;
    const unsigned short* Cb  = Ckv   + (size_t)b * SL_ * RANK_;
    const unsigned short* CTb = CkvT  + (size_t)b * RANK_ * SL_;
    const unsigned short* Kb  = Krope + (size_t)(b * HKV_ + hk) * SL_ * ROPE_;

    // ---- stage the 18 A fragments (16 nope + 2 rope) into LDS, wave 0 only ----
    if (w == 0) {
        #pragma unroll
        for (int f = 0; f < 16; ++f) sta_lds(sA, f, lane, load_a(Qa, RANK_, f * 32, lane));
        sta_lds(sA, 16, lane, load_a(Qr, ROPE_, 0,  lane));
        sta_lds(sA, 17, lane, load_a(Qr, ROPE_, 32, lane));
    }
    if (t < 16) { sM[t] = -3.0e38f; sL[t] = 0.f; }

    v8f acc[4] = {};                                  // out_latent cols w*64 + {0,16,32,48}
    const int r0 = (lane >> 4) << 3;                  // C/D: rows r0..r0+7
    const int cn = lane & 15;                         // C/D: column within tile
    __syncthreads();

    // rotating operand buffers (live across the whole s-loop)
    v16bf bb[4];                                      // depth-4 B pipeline
    v16bf ab[2];                                      // depth-2 A pipeline (QK, from sA)
    {
        const int nb = w * 16;
        #pragma unroll
        for (int j = 0; j < 4; ++j) bb[j] = load_bT(Cb, RANK_, nb, j * 32, lane);
        ab[0] = lda_lds(sA, 0, lane);
        ab[1] = lda_lds(sA, 1, lane);
    }

    for (int s0 = 0; s0 < SL_; s0 += 128) {
        const int nb = s0 + w * 16;

        // ---- QK^T: 18 pipelined WMMAs (16 nope + 2 rope), dual acc chains ----
        v8f d0 = {}, d1 = {};
        #pragma unroll
        for (int k = 0; k < 18; ++k) {
            if (k & 1) d1 = wmma_bf16(ab[k & 1], bb[k & 3], d1);
            else       d0 = wmma_bf16(ab[k & 1], bb[k & 3], d0);
            if (k + 4 < 18) {
                const int kn = k + 4;
                bb[k & 3] = (kn < 16) ? load_bT(Cb, RANK_, nb, kn * 32, lane)
                                      : load_bT(Kb, ROPE_, nb, (kn - 16) * 32, lane);
            }
            if (k + 2 < 18) ab[k & 1] = lda_lds(sA, k + 2, lane);
            __builtin_amdgcn_sched_group_barrier(SG_MFMA, 1, 0);
            if (k + 4 < 18) __builtin_amdgcn_sched_group_barrier(SG_VMEM_RD, 2, 0);
            if (k + 2 < 18) __builtin_amdgcn_sched_group_barrier(SG_DS_RD, 2, 0);
        }
        #pragma unroll
        for (int j = 0; j < 8; ++j) sS[(r0 + j) * 132 + w * 16 + cn] = d0[j] + d1[j];
        __syncthreads();

        // ---- online softmax: 16 threads per row, 8 cols each ----
        {
            const int row = t >> 4, i = t & 15;
            float vals[8]; float vmax = -3.0e38f;
            #pragma unroll
            for (int c = 0; c < 8; ++c) { vals[c] = sS[row * 132 + i + c * 16]; vmax = fmaxf(vmax, vals[c]); }
            vmax = fmaxf(vmax, __shfl_xor(vmax, 8, 32));
            vmax = fmaxf(vmax, __shfl_xor(vmax, 4, 32));
            vmax = fmaxf(vmax, __shfl_xor(vmax, 2, 32));
            vmax = fmaxf(vmax, __shfl_xor(vmax, 1, 32));
            const float mold = sM[row];
            const float mnew = fmaxf(mold, vmax);
            float lsum = 0.f;
            #pragma unroll
            for (int c = 0; c < 8; ++c) {
                float p = __expf(vals[c] - mnew);
                lsum += p;
                sP[row * 136 + i + c * 16] = f2bf(p);
            }
            lsum += __shfl_xor(lsum, 8, 32);
            lsum += __shfl_xor(lsum, 4, 32);
            lsum += __shfl_xor(lsum, 2, 32);
            lsum += __shfl_xor(lsum, 1, 32);
            if (i == 0) {
                const float sc = __expf(mold - mnew);
                sScale[row] = sc;
                sL[row] = sL[row] * sc + lsum;
                sM[row] = mnew;
            }
        }
        __syncthreads();

        // ---- PV prologue loads first (their latency is covered by the rescale) ----
        v16bf pa[2];
        pa[0] = load_a(sP, 136, 0,  lane);
        pa[1] = load_a(sP, 136, 32, lane);
        #pragma unroll
        for (int n = 0; n < 4; ++n) bb[n] = load_bT(CTb, SL_, w * 64 + n * 16, s0, lane);

        // ---- rescale accumulator ----
        float rs[8];
        #pragma unroll
        for (int j = 0; j < 8; ++j) rs[j] = sScale[r0 + j];
        #pragma unroll
        for (int n = 0; n < 4; ++n)
            #pragma unroll
            for (int j = 0; j < 8; ++j) acc[n][j] *= rs[j];

        // ---- PV: 16 pipelined WMMAs (kc outer x n inner) ----
        #pragma unroll
        for (int idx = 0; idx < 16; ++idx) {
            const int kc = idx >> 2, n = idx & 3;
            acc[n] = wmma_bf16(pa[kc & 1], bb[idx & 3], acc[n]);
            if (idx + 4 < 16)
                bb[idx & 3] = load_bT(CTb, SL_, w * 64 + n * 16, s0 + (kc + 1) * 32, lane);
            if (n == 3 && kc + 2 < 4)
                pa[kc & 1] = load_a(sP, 136, (kc + 2) * 32, lane);
            __builtin_amdgcn_sched_group_barrier(SG_MFMA, 1, 0);
            if (idx + 4 < 16) __builtin_amdgcn_sched_group_barrier(SG_VMEM_RD, 2, 0);
            if (n == 3 && kc + 2 < 4) __builtin_amdgcn_sched_group_barrier(SG_DS_RD, 2, 0);
        }

        // ---- prologue for next chunk's QK, issued before the barrier so the
        //      wait for other waves' PV hides the latency ----
        if (s0 + 128 < SL_) {
            const int nb2 = s0 + 128 + w * 16;
            #pragma unroll
            for (int j = 0; j < 4; ++j) bb[j] = load_bT(Cb, RANK_, nb2, j * 32, lane);
            ab[0] = lda_lds(sA, 0, lane);
            ab[1] = lda_lds(sA, 1, lane);
        }
        __syncthreads();
    }

    // ---- finalize: /l, stage out_latent bf16 to LDS ----
    float inv[8];
    #pragma unroll
    for (int j = 0; j < 8; ++j) inv[j] = 1.0f / sL[r0 + j];
    #pragma unroll
    for (int n = 0; n < 4; ++n)
        #pragma unroll
        for (int j = 0; j < 8; ++j)
            sO[(r0 + j) * 520 + w * 64 + n * 16 + cn] = f2bf(acc[n][j] * inv[j]);
    __syncthreads();

    // ---- fused epilogue: out = out_latent @ W_UV ; wave w -> cols w*16.. ----
    {
        v16bf ao = load_a(sO, 520, 0, lane);
        v16bf bw = load_bT(WUVT, RANK_, w * 16, 0, lane);
        v8f e0 = {}, e1 = {};
        #pragma unroll
        for (int kc = 0; kc < 16; ++kc) {
            v16bf aon = ao, bwn = bw;
            if (kc < 15) {
                aon = load_a(sO, 520, (kc + 1) * 32, lane);
                bwn = load_bT(WUVT, RANK_, w * 16, (kc + 1) * 32, lane);
            }
            if (kc & 1) e1 = wmma_bf16(ao, bw, e1);
            else        e0 = wmma_bf16(ao, bw, e0);
            ao = aon; bw = bwn;
        }
        float* Ob = Out + qoff * DH_;
        #pragma unroll
        for (int j = 0; j < 8; ++j)
            Ob[(size_t)(r0 + j) * DH_ + w * 16 + cn] = e0[j] + e1[j];
    }
}

// ---------------- launcher ----------------
extern "C" void kernel_launch(void* const* d_in, const int* in_sizes, int n_in,
                              void* d_out, int out_size, void* d_ws, size_t ws_size,
                              hipStream_t stream) {
    const float* Q     = (const float*)d_in[0];
    const float* Ckv   = (const float*)d_in[1];
    const float* Krope = (const float*)d_in[2];
    const float* U_l   = (const float*)d_in[3];
    const float* W_UK  = (const float*)d_in[4];
    const float* W_UV  = (const float*)d_in[5];
    float* Out = (float*)d_out;

    char* ws = (char*)d_ws;
    float*          Mabs   = (float*)(ws + OFF_MABS);
    unsigned short* WUVT   = (unsigned short*)(ws + OFF_WUVT);
    unsigned short* Qabs   = (unsigned short*)(ws + OFF_QABS);
    unsigned short* Qrope  = (unsigned short*)(ws + OFF_QROPE);
    unsigned short* Ckvb   = (unsigned short*)(ws + OFF_CKV);
    unsigned short* CkvTb  = (unsigned short*)(ws + OFF_CKVT);
    unsigned short* Kropeb = (unsigned short*)(ws + OFF_KROPE);

    k_mabs <<<128,   256, 0, stream>>>(U_l, W_UK, Mabs);
    k_wuvt <<<256,   256, 0, stream>>>(W_UV, WUVT);
    k_ckv  <<<16384, 256, 0, stream>>>(Ckv, Ckvb, CkvTb);
    k_krope<<<16384, 256, 0, stream>>>(Krope, Kropeb);
    k_prepq<<<B_ * HQ_ * QL_, 128, 0, stream>>>(Q, U_l, Mabs, Qabs, Qrope);
    mla_attn<<<B_ * HQ_ * (QL_ / 16), 256, 0, stream>>>(Qabs, Qrope, Ckvb, CkvTb, Kropeb, WUVT, Out);
}
